// MS2DGBlock_61117384622239
// MI455X (gfx1250) — matrix operand order repaired and checked by reading, not compiled, not run on hardware
//
#include <hip/hip_runtime.h>
#include <hip/hip_bf16.h>
#include <math.h>

typedef __attribute__((ext_vector_type(16))) _Float16 v16h;
typedef __attribute__((ext_vector_type(8)))  float    v8f;

#define TPB 256

// =====================================================================
// Batched GEMM, f32 in/out, f16 WMMA core (v_wmma_f32_16x16x32_f16),
// LDS-staged 64x64 block tiles, 256 threads (8 waves), wave32.
//   Out[b][m][n] = sum_k A[b][m][k] * B[b][k][n] (+ bias[m])
//   TA: A element read as A[k*lda + m]; TB: B element read as B[n*ldb + k]
// Wave w: m-tile = (w>>1)*16, n-pair = (w&1)*32 -> two 16x16 WMMA tiles.
// =====================================================================
#define LDSPAD 40   // halves per row: 80B rows -> 16B-aligned rows, conflict-stagger

template<int TA, int TB>
__global__ __launch_bounds__(256) void gemm_f16wmma_kernel(
    const float* __restrict__ A, const float* __restrict__ Bm,
    const float* __restrict__ bias, float* __restrict__ C,
    int M, int K, int N, int lda, int ldb, int ldc,
    long sA, long sB, long sC)
{
  __shared__ _Float16 As[64][LDSPAD];   // [m][k]
  __shared__ _Float16 Bs[64][LDSPAD];   // [n][k]

  const int t = threadIdx.x;
  const int lane = t & 31;
  const int wave = t >> 5;
  const int g = lane >> 4;          // lane half-group
  const int r = lane & 15;
  const int mt = (wave >> 1) * 16;  // wave's m offset in block tile
  const int np = (wave & 1) * 32;   // wave's n offset in block tile
  const int m0b = blockIdx.y * 64;
  const int n0b = blockIdx.x * 64;
  const int b = blockIdx.z;
  const float* Ab = A + (long)b * sA;
  const float* Bb = Bm + (long)b * sB;
  float* Cb = C + (long)b * sC;

  v8f acc0 = {}, acc1 = {};

  for (int k0 = 0; k0 < K; k0 += 32) {
    // ---- cooperative load: A tile 64(m) x 32(k) ----
    if (TA == 0) {                      // A[m][k], contiguous k
      int m = t >> 2, kb = (t & 3) * 8;
      int gm = m0b + m, kg = k0 + kb;
      if (gm < M && kg + 7 < K) {
        const float4* p = (const float4*)(Ab + (long)gm * lda + kg);
        float4 v0 = p[0], v1 = p[1];
        __builtin_prefetch(p + 8, 0, 1);              // next K tile
        As[m][kb + 0] = (_Float16)v0.x; As[m][kb + 1] = (_Float16)v0.y;
        As[m][kb + 2] = (_Float16)v0.z; As[m][kb + 3] = (_Float16)v0.w;
        As[m][kb + 4] = (_Float16)v1.x; As[m][kb + 5] = (_Float16)v1.y;
        As[m][kb + 6] = (_Float16)v1.z; As[m][kb + 7] = (_Float16)v1.w;
      } else {
#pragma unroll
        for (int j = 0; j < 8; ++j) {
          float v = 0.f;
          if (gm < M && kg + j < K) v = Ab[(long)gm * lda + kg + j];
          As[m][kb + j] = (_Float16)v;
        }
      }
    } else {                            // A[k][m], contiguous m (Gram)
      int k = t >> 3, mb = (t & 7) * 8;
      int kg = k0 + k, gm = m0b + mb;
      if (kg < K && gm + 7 < M) {
        const float4* p = (const float4*)(Ab + (long)kg * lda + gm);
        float4 v0 = p[0], v1 = p[1];
        As[mb + 0][k] = (_Float16)v0.x; As[mb + 1][k] = (_Float16)v0.y;
        As[mb + 2][k] = (_Float16)v0.z; As[mb + 3][k] = (_Float16)v0.w;
        As[mb + 4][k] = (_Float16)v1.x; As[mb + 5][k] = (_Float16)v1.y;
        As[mb + 6][k] = (_Float16)v1.z; As[mb + 7][k] = (_Float16)v1.w;
      } else {
#pragma unroll
        for (int j = 0; j < 8; ++j) {
          float v = 0.f;
          if (kg < K && gm + j < M) v = Ab[(long)kg * lda + gm + j];
          As[mb + j][k] = (_Float16)v;
        }
      }
    }
    // ---- cooperative load: B tile 32(k) x 64(n), stored n-major ----
    if (TB == 0) {                      // B[k][n], contiguous n
      int k = t >> 3, nb = (t & 7) * 8;
      int kg = k0 + k, gn = n0b + nb;
      if (kg < K && gn + 7 < N) {
        const float4* p = (const float4*)(Bb + (long)kg * ldb + gn);
        float4 v0 = p[0], v1 = p[1];
        Bs[nb + 0][k] = (_Float16)v0.x; Bs[nb + 1][k] = (_Float16)v0.y;
        Bs[nb + 2][k] = (_Float16)v0.z; Bs[nb + 3][k] = (_Float16)v0.w;
        Bs[nb + 4][k] = (_Float16)v1.x; Bs[nb + 5][k] = (_Float16)v1.y;
        Bs[nb + 6][k] = (_Float16)v1.z; Bs[nb + 7][k] = (_Float16)v1.w;
      } else {
#pragma unroll
        for (int j = 0; j < 8; ++j) {
          float v = 0.f;
          if (kg < K && gn + j < N) v = Bb[(long)kg * ldb + gn + j];
          Bs[nb + j][k] = (_Float16)v;
        }
      }
    } else {                            // B[n][k], contiguous k (diff_pool)
      int n = t >> 2, kb = (t & 3) * 8;
      int gn = n0b + n, kg = k0 + kb;
      if (gn < N && kg + 7 < K) {
        const float4* p = (const float4*)(Bb + (long)gn * ldb + kg);
        float4 v0 = p[0], v1 = p[1];
        __builtin_prefetch(p + 8, 0, 1);
        Bs[n][kb + 0] = (_Float16)v0.x; Bs[n][kb + 1] = (_Float16)v0.y;
        Bs[n][kb + 2] = (_Float16)v0.z; Bs[n][kb + 3] = (_Float16)v0.w;
        Bs[n][kb + 4] = (_Float16)v1.x; Bs[n][kb + 5] = (_Float16)v1.y;
        Bs[n][kb + 6] = (_Float16)v1.z; Bs[n][kb + 7] = (_Float16)v1.w;
      } else {
#pragma unroll
        for (int j = 0; j < 8; ++j) {
          float v = 0.f;
          if (gn < N && kg + j < K) v = Bb[(long)gn * ldb + kg + j];
          Bs[n][kb + j] = (_Float16)v;
        }
      }
    }
    __syncthreads();

    // ---- fragments from LDS (packed b32 reads) ----
    union U { v16h v; unsigned u[8]; };
    U af, bf0, bf1;
#pragma unroll
    for (int j = 0; j < 8; ++j) {
      int i2 = 2 * j;
      // A 16x32 f16 layout: lanes 0-15 K{0..7,16..23}, lanes 16-31 K{8..15,24..31}
      int ka = i2 + (i2 < 8 ? 0 : 8) + g * 8;
      af.u[j] = *(const unsigned*)&As[mt + r][ka];
      // B 32x16 f16 layout: lanes 0-15 K{0..7,16..23}, lanes 16-31 K{8..15,24..31}
      int kbx = i2 + (i2 < 8 ? 0 : 8) + g * 8;
      bf0.u[j] = *(const unsigned*)&Bs[np + r][kbx];
      bf1.u[j] = *(const unsigned*)&Bs[np + 16 + r][kbx];
    }
    acc0 = __builtin_amdgcn_wmma_f32_16x16x32_f16(false, af.v, false, bf0.v,
                                                  (short)0, acc0, false, false);
    acc1 = __builtin_amdgcn_wmma_f32_16x16x32_f16(false, af.v, false, bf1.v,
                                                  (short)0, acc1, false, false);
    __syncthreads();
  }

  // ---- store D: vgpr rr, lanes 0-15 -> M=rr, lanes 16-31 -> M=rr+8 ----
#pragma unroll
  for (int rr = 0; rr < 8; ++rr) {
    int m = m0b + mt + rr + g * 8;
    if (m < M) {
      float bv = bias ? bias[m] : 0.f;
      int n = n0b + np + r;
      if (n < N)      Cb[(long)m * ldc + n] = acc0[rr] + bv;
      if (n + 16 < N) Cb[(long)m * ldc + n + 16] = acc1[rr] + bv;
    }
  }
}

// =====================================================================
// Normalization kernels (instnorm per (b,c); bnorm per channel)
// =====================================================================
__global__ void inst_stats_kernel(const float* __restrict__ x, float* mean, float* rstd,
                                  int S, float eps)
{
  int bc = blockIdx.x;
  const float* p = x + (long)bc * S;
  float s = 0.f, s2 = 0.f;
  for (int i = threadIdx.x; i < S; i += blockDim.x) { float v = p[i]; s += v; s2 += v * v; }
  __shared__ float sa[TPB], sb[TPB];
  sa[threadIdx.x] = s; sb[threadIdx.x] = s2; __syncthreads();
  for (int o = TPB / 2; o > 0; o >>= 1) {
    if (threadIdx.x < (unsigned)o) { sa[threadIdx.x] += sa[threadIdx.x + o]; sb[threadIdx.x] += sb[threadIdx.x + o]; }
    __syncthreads();
  }
  if (threadIdx.x == 0) {
    float m = sa[0] / S; float var = sb[0] / S - m * m;
    mean[bc] = m; rstd[bc] = rsqrtf(var + eps);
  }
}

__global__ void chan_stats_kernel(const float* __restrict__ x,
                                  const float* __restrict__ imean, const float* __restrict__ irstd,
                                  int Bn, int C, int S, float* mean, float* rstd, float eps)
{
  int c = blockIdx.x;
  long total = (long)Bn * S;
  float s = 0.f, s2 = 0.f;
  for (long t = threadIdx.x; t < total; t += blockDim.x) {
    int b = (int)(t / S); int i = (int)(t % S);
    float v = x[((long)b * C + c) * S + i];
    if (imean) v = (v - imean[b * C + c]) * irstd[b * C + c];
    s += v; s2 += v * v;
  }
  __shared__ float sa[TPB], sb[TPB];
  sa[threadIdx.x] = s; sb[threadIdx.x] = s2; __syncthreads();
  for (int o = TPB / 2; o > 0; o >>= 1) {
    if (threadIdx.x < (unsigned)o) { sa[threadIdx.x] += sa[threadIdx.x + o]; sb[threadIdx.x] += sb[threadIdx.x + o]; }
    __syncthreads();
  }
  if (threadIdx.x == 0) {
    float m = sa[0] / (float)total; float var = sb[0] / (float)total - m * m;
    mean[c] = m; rstd[c] = rsqrtf(var + eps);
  }
}

__global__ void norm_apply_kernel(const float* __restrict__ x, float* __restrict__ y,
                                  const float* __restrict__ imean, const float* __restrict__ irstd,
                                  const float* __restrict__ cmean, const float* __restrict__ crstd,
                                  const float* __restrict__ g, const float* __restrict__ bb,
                                  int C, int S, int relu, long total)
{
  long i = (long)blockIdx.x * blockDim.x + threadIdx.x;
  if (i >= total) return;
  long bc = i / S;
  int c = (int)(bc % C);
  float v = x[i];
  if (imean) v = (v - imean[bc]) * irstd[bc];
  v = (v - cmean[c]) * crstd[c] * g[c] + bb[c];
  if (relu) v = fmaxf(v, 0.f);
  y[i] = v;
}

// =====================================================================
// kNN / graph feature
// =====================================================================
__global__ void sumsq_kernel(const float* __restrict__ x, float* __restrict__ xx,
                             int C, int N, long total)
{
  long i = (long)blockIdx.x * blockDim.x + threadIdx.x;
  if (i >= total) return;
  long b = i / N; int n = (int)(i % N);
  float s = 0.f;
  for (int c = 0; c < C; ++c) { float v = x[((long)b * C + c) * N + n]; s += v * v; }
  xx[i] = s;
}

#define KNN_K 20
__global__ void topk_kernel(const float* __restrict__ G, const float* __restrict__ xx,
                            int* __restrict__ idx_out, int N)
{
  int n = blockIdx.x * blockDim.x + threadIdx.x;
  if (n >= N) return;
  float best[KNN_K]; int bidx[KNN_K];
#pragma unroll
  for (int j = 0; j < KNN_K; ++j) { best[j] = -3.4e38f; bidx[j] = 0; }
  float xn = xx[n];
  const float* row = G + (long)n * N;
  for (int m = 0; m < N; ++m) {
    float d = 2.f * row[m] - xn - xx[m];
    if (d > best[KNN_K - 1]) {
      int j = KNN_K - 1;
      while (j > 0 && best[j - 1] < d) { best[j] = best[j - 1]; bidx[j] = bidx[j - 1]; --j; }
      best[j] = d; bidx[j] = m;
    }
  }
#pragma unroll
  for (int j = 0; j < KNN_K; ++j) idx_out[(long)n * KNN_K + j] = bidx[j];
}

__global__ void graph_feat_kernel(const float* __restrict__ x, const int* __restrict__ idx,
                                  float* __restrict__ feat, int C, int N, int Kk, long total)
{
  long i = (long)blockIdx.x * blockDim.x + threadIdx.x;
  if (i >= total) return;
  int kk = (int)(i % Kk); long rest = i / Kk;
  int n = (int)(rest % N); rest /= N;
  int ch = (int)(rest % (2 * C)); long b = rest / (2 * C);
  float v;
  if (ch < C) {
    v = x[((long)b * C + ch) * N + n];
  } else {
    int cc = ch - C;
    float xc = x[((long)b * C + cc) * N + n];
    int j = idx[((long)b * N + n) * Kk + kk];
    v = xc - x[((long)b * C + cc) * N + j];
  }
  feat[i] = v;
}

// =====================================================================
// Attention (softmax over k of q*k, weighted sum of v) fused
// =====================================================================
__global__ void attn_kernel(const float* __restrict__ q, const float* __restrict__ k,
                            const float* __restrict__ v, const float* __restrict__ x,
                            const float* __restrict__ gamma, float* __restrict__ out,
                            int Kk, long total)
{
  long i = (long)blockIdx.x * blockDim.x + threadIdx.x;
  if (i >= total) return;
  const float* qp = q + i * Kk;
  const float* kp = k + i * Kk;
  const float* vp = v + i * Kk;
  float mx = -3.4e38f;
  for (int t = 0; t < Kk; ++t) mx = fmaxf(mx, qp[t] * kp[t]);
  float sum = 0.f, acc = 0.f;
  for (int t = 0; t < Kk; ++t) {
    float e = expf(qp[t] * kp[t] - mx);
    sum += e; acc += e * vp[t];
  }
  out[i] = x[i] + gamma[0] * (acc / sum);
}

__global__ void maxk_kernel(const float* __restrict__ in, float* __restrict__ out,
                            int Kk, long total)
{
  long i = (long)blockIdx.x * blockDim.x + threadIdx.x;
  if (i >= total) return;
  const float* p = in + i * Kk;
  float m = p[0];
  for (int t = 1; t < Kk; ++t) m = fmaxf(m, p[t]);
  out[i] = m;
}

// =====================================================================
// Elementwise helpers
// =====================================================================
__global__ void add_kernel(float* __restrict__ out, const float* __restrict__ a,
                           const float* __restrict__ b, long total)
{
  long i = (long)blockIdx.x * blockDim.x + threadIdx.x;
  if (i >= total) return;
  out[i] = a[i] + b[i];
}

__global__ void copy_chan_kernel(float* __restrict__ dst, const float* __restrict__ src,
                                 int Cdst, int Csrc, int c0, int S, long total)
{
  long i = (long)blockIdx.x * blockDim.x + threadIdx.x;
  if (i >= total) return;
  long b = i / ((long)Csrc * S);
  long rem = i % ((long)Csrc * S);
  int c = (int)(rem / S); int ss = (int)(rem % S);
  dst[((long)b * Cdst + c0 + c) * S + ss] = src[i];
}

__global__ void transpose_kernel(const float* __restrict__ src, float* __restrict__ dst,
                                 int C, int S, long total)
{
  long i = (long)blockIdx.x * blockDim.x + threadIdx.x;
  if (i >= total) return;
  long b = i / ((long)C * S);
  long rem = i % ((long)C * S);
  int c = (int)(rem / S); int ss = (int)(rem % S);
  dst[((long)b * S + ss) * C + c] = src[i];
}

// softmax over contiguous last dim (one block / row), in-place
__global__ void softmax_row_kernel(float* __restrict__ x, int L)
{
  float* p = x + (long)blockIdx.x * L;
  __shared__ float sh[TPB];
  float mx = -3.4e38f;
  for (int i = threadIdx.x; i < L; i += TPB) mx = fmaxf(mx, p[i]);
  sh[threadIdx.x] = mx; __syncthreads();
  for (int o = TPB / 2; o > 0; o >>= 1) {
    if (threadIdx.x < (unsigned)o) sh[threadIdx.x] = fmaxf(sh[threadIdx.x], sh[threadIdx.x + o]);
    __syncthreads();
  }
  mx = sh[0]; __syncthreads();
  float s = 0.f;
  for (int i = threadIdx.x; i < L; i += TPB) s += expf(p[i] - mx);
  sh[threadIdx.x] = s; __syncthreads();
  for (int o = TPB / 2; o > 0; o >>= 1) {
    if (threadIdx.x < (unsigned)o) sh[threadIdx.x] += sh[threadIdx.x + o];
    __syncthreads();
  }
  float inv = 1.f / sh[0];
  for (int i = threadIdx.x; i < L; i += TPB) p[i] = expf(p[i] - mx) * inv;
}

// softmax over channel dim (stride S), one thread per (b, s), in-place
__global__ void softmax_chan_kernel(float* __restrict__ x, int C, int S, long total)
{
  long i = (long)blockIdx.x * blockDim.x + threadIdx.x;
  if (i >= total) return;
  long b = i / S; int ss = (int)(i % S);
  float* p = x + (long)b * C * S + ss;
  float mx = -3.4e38f;
  for (int c = 0; c < C; ++c) mx = fmaxf(mx, p[(long)c * S]);
  float s = 0.f;
  for (int c = 0; c < C; ++c) s += expf(p[(long)c * S] - mx);
  float inv = 1.f / s;
  for (int c = 0; c < C; ++c) p[(long)c * S] = expf(p[(long)c * S] - mx) * inv;
}

// logits = w . o[:, :, n] + b  (Cin=128)
__global__ void logits_kernel(const float* __restrict__ o, const float* __restrict__ w,
                              const float* __restrict__ bias, float* __restrict__ out,
                              int C, int N, long total)
{
  long i = (long)blockIdx.x * blockDim.x + threadIdx.x;
  if (i >= total) return;
  long b = i / N; int n = (int)(i % N);
  float s = bias[0];
  for (int c = 0; c < C; ++c) s += w[c] * o[((long)b * C + c) * N + n];
  out[i] = s;
}

// =====================================================================
// weighted_8points: XwX accumulation + 9x9 Jacobi eigh + episym residual
// =====================================================================
__global__ void xwx_kernel(const float* __restrict__ xs, const float* __restrict__ logits,
                           float* __restrict__ xwx, int N)
{
  int b = blockIdx.x;
  __shared__ float sh[81];
  if (threadIdx.x < 81) sh[threadIdx.x] = 0.f;
  __syncthreads();
  float part[45];
#pragma unroll
  for (int t = 0; t < 45; ++t) part[t] = 0.f;
  for (int n = threadIdx.x; n < N; n += blockDim.x) {
    const float* p = xs + ((long)b * N + n) * 4;
    float x0 = p[0], x1 = p[1], x2 = p[2], x3 = p[3];
    float X[9] = { x2 * x0, x2 * x1, x2, x3 * x0, x3 * x1, x3, x0, x1, 1.f };
    float w = tanhf(logits[(long)b * N + n]); w = fmaxf(w, 0.f);
    int t = 0;
    for (int pp = 0; pp < 9; ++pp)
      for (int qq = pp; qq < 9; ++qq)
        part[t++] += w * X[pp] * X[qq];
  }
  {
    int t = 0;
    for (int pp = 0; pp < 9; ++pp)
      for (int qq = pp; qq < 9; ++qq)
        atomicAdd(&sh[pp * 9 + qq], part[t++]);
  }
  __syncthreads();
  if (threadIdx.x < 81) {
    int pp = threadIdx.x / 9, qq = threadIdx.x % 9;
    float v = (pp <= qq) ? sh[pp * 9 + qq] : sh[qq * 9 + pp];
    xwx[(long)b * 81 + threadIdx.x] = v;
  }
}

__global__ void eigh9_kernel(const float* __restrict__ xwx, float* __restrict__ e, int Bn)
{
  int b = blockIdx.x * blockDim.x + threadIdx.x;
  if (b >= Bn) return;
  float A[81], V[81];
  for (int i = 0; i < 81; ++i) { A[i] = xwx[(long)b * 81 + i]; V[i] = 0.f; }
  for (int i = 0; i < 9; ++i) V[i * 9 + i] = 1.f;
  for (int sweep = 0; sweep < 40; ++sweep) {
    for (int p = 0; p < 8; ++p) {
      for (int q = p + 1; q < 9; ++q) {
        float apq = A[p * 9 + q];
        if (fabsf(apq) < 1e-14f) continue;
        float app = A[p * 9 + p], aqq = A[q * 9 + q];
        float tau = (aqq - app) / (2.f * apq);
        float t = ((tau >= 0.f) ? 1.f : -1.f) / (fabsf(tau) + sqrtf(1.f + tau * tau));
        float c = rsqrtf(1.f + t * t);
        float s = t * c;
        for (int i = 0; i < 9; ++i) {
          float aip = A[i * 9 + p], aiq = A[i * 9 + q];
          A[i * 9 + p] = c * aip - s * aiq;
          A[i * 9 + q] = s * aip + c * aiq;
        }
        for (int i = 0; i < 9; ++i) {
          float api = A[p * 9 + i], aqi = A[q * 9 + i];
          A[p * 9 + i] = c * api - s * aqi;
          A[q * 9 + i] = s * api + c * aqi;
        }
        for (int i = 0; i < 9; ++i) {
          float vip = V[i * 9 + p], viq = V[i * 9 + q];
          V[i * 9 + p] = c * vip - s * viq;
          V[i * 9 + q] = s * vip + c * viq;
        }
      }
    }
  }
  int jmin = 0; float dmin = A[0];
  for (int j = 1; j < 9; ++j) { float d = A[j * 9 + j]; if (d < dmin) { dmin = d; jmin = j; } }
  float nrm = 0.f;
  for (int i = 0; i < 9; ++i) { float v = V[i * 9 + jmin]; nrm += v * v; }
  nrm = rsqrtf(nrm + 1e-30f);
  for (int i = 0; i < 9; ++i) e[(long)b * 9 + i] = V[i * 9 + jmin] * nrm;
}

__global__ void episym_kernel(const float* __restrict__ xs, const float* __restrict__ e,
                              float* __restrict__ res, int N, long total)
{
  long i = (long)blockIdx.x * blockDim.x + threadIdx.x;
  if (i >= total) return;
  long b = i / N; int n = (int)(i % N);
  const float* F = e + b * 9;
  const float* p = xs + ((long)b * N + n) * 4;
  float x0 = p[0], y0 = p[1], x1 = p[2], y1 = p[3];
  float Fx0 = F[0] * x0 + F[1] * y0 + F[2];
  float Fx1 = F[3] * x0 + F[4] * y0 + F[5];
  float Fx2 = F[6] * x0 + F[7] * y0 + F[8];
  float Ft0 = F[0] * x1 + F[3] * y1 + F[6];
  float Ft1 = F[1] * x1 + F[4] * y1 + F[7];
  float x2Fx1 = x1 * Fx0 + y1 * Fx1 + Fx2;
  float eps = 1e-15f;
  float r = x2Fx1 * x2Fx1 * (1.f / (Fx0 * Fx0 + Fx1 * Fx1 + eps) +
                             1.f / (Ft0 * Ft0 + Ft1 * Ft1 + eps));
  res[i] = r;
}

// =====================================================================
// Host-side wiring
// =====================================================================
struct Conv { const float* b; const float* w; };
struct BN   { const float* b; const float* g; };
struct CNp  { BN bn1, bn2; Conv c1, c2, sc; int has_sc; };
struct OAp  { BN bn1, bn2, bn3; Conv c1, c2, c3; };
struct Stats { float* imean; float* irstd; float* cmean; float* crstd; };

static inline dim3 grid1d(long total) { return dim3((unsigned)((total + TPB - 1) / TPB)); }

static void gemm(const float* A, const float* B, const float* bias, float* C,
                 int M, int K, int N, int lda, int ldb, int ldc,
                 long sA, long sB, long sC, int batch, int tA, int tB, hipStream_t st)
{
  dim3 grid((N + 63) / 64, (M + 63) / 64, batch);
  if (tA)
    gemm_f16wmma_kernel<1, 0><<<grid, 256, 0, st>>>(A, B, bias, C, M, K, N, lda, ldb, ldc, sA, sB, sC);
  else if (tB)
    gemm_f16wmma_kernel<0, 1><<<grid, 256, 0, st>>>(A, B, bias, C, M, K, N, lda, ldb, ldc, sA, sB, sC);
  else
    gemm_f16wmma_kernel<0, 0><<<grid, 256, 0, st>>>(A, B, bias, C, M, K, N, lda, ldb, ldc, sA, sB, sC);
}

static void norm_relu(const float* x, float* y, const float* g, const float* bb,
                      int Bn, int C, int S, int do_inst, int relu, Stats st, hipStream_t s)
{
  if (do_inst)
    inst_stats_kernel<<<Bn * C, TPB, 0, s>>>(x, st.imean, st.irstd, S, 1e-3f);
  chan_stats_kernel<<<C, TPB, 0, s>>>(x, do_inst ? st.imean : nullptr,
                                      do_inst ? st.irstd : nullptr, Bn, C, S,
                                      st.cmean, st.crstd, 1e-5f);
  long total = (long)Bn * C * S;
  norm_apply_kernel<<<grid1d(total), TPB, 0, s>>>(x, y, do_inst ? st.imean : nullptr,
                                                  do_inst ? st.irstd : nullptr,
                                                  st.cmean, st.crstd, g, bb, C, S, relu, total);
}

static void graph_feature(const float* x, float* feat, float* xsq, float* G, int* idx,
                          int Bn, int C, int N, int Kk, hipStream_t s)
{
  long tn = (long)Bn * N;
  sumsq_kernel<<<grid1d(tn), TPB, 0, s>>>(x, xsq, C, N, tn);
  for (int b = 0; b < Bn; ++b) {
    const float* xb = x + (long)b * C * N;
    gemm(xb, xb, nullptr, G, N, C, N, /*lda*/N, /*ldb*/N, /*ldc*/N, 0, 0, 0, 1,
         /*tA*/1, /*tB*/0, s);                 // Gram = X^T X
    topk_kernel<<<grid1d(N), TPB, 0, s>>>(G, xsq + (long)b * N, idx + (long)b * N * Kk, N);
  }
  long tf = (long)Bn * 2 * C * N * Kk;
  graph_feat_kernel<<<grid1d(tf), TPB, 0, s>>>(x, idx, feat, C, N, Kk, tf);
}

static void context_norm_layer(const float* hin, float* hout, int Cin, const CNp& p,
                               float* t, float* h1, float* h2, float* hsc,
                               Stats st, int Bn, int S, hipStream_t s)
{
  norm_relu(hin, t, p.bn1.g, p.bn1.b, Bn, Cin, S, 1, 1, st, s);
  gemm(p.c1.w, t, p.c1.b, h1, 128, Cin, S, Cin, S, S, 0, (long)Cin * S, (long)128 * S, Bn, 0, 0, s);
  norm_relu(h1, h1, p.bn2.g, p.bn2.b, Bn, 128, S, 1, 1, st, s);
  gemm(p.c2.w, h1, p.c2.b, h2, 128, 128, S, 128, S, S, 0, (long)128 * S, (long)128 * S, Bn, 0, 0, s);
  const float* scv = hin;
  if (p.has_sc) {
    gemm(p.sc.w, hin, p.sc.b, hsc, 128, Cin, S, Cin, S, S, 0, (long)Cin * S, (long)128 * S, Bn, 0, 0, s);
    scv = hsc;
  }
  long tot = (long)Bn * 128 * S;
  add_kernel<<<grid1d(tot), TPB, 0, s>>>(hout, h2, scv, tot);
}

extern "C" void kernel_launch(void* const* d_in, const int* in_sizes, int n_in,
                              void* d_out, int out_size, void* d_ws, size_t ws_size,
                              hipStream_t stream)
{
  (void)in_sizes; (void)n_in; (void)out_size; (void)ws_size;
  const int Bn = 8, N = 2000, HALF = 64, C = 128, CL = 500, Kk = KNN_K;
  const int NSP = N * Kk;               // 40000
  hipStream_t s = stream;

  // ---------------- inputs (data, xs, then params leaves in pytree order) ----
  const float* data = (const float*)d_in[0];   // [B,4,N]
  const float* xs   = (const float*)d_in[1];   // [B,N,4]
  int cur = 2;
  auto nxt = [&]() { return (const float*)d_in[cur++]; };
  auto rdBN = [&]() { BN r; r.b = nxt(); r.g = nxt(); return r; };
  auto rdCv = [&]() { Conv r; r.b = nxt(); r.w = nxt(); return r; };

  // params['att'] : bn, c, gamma, k, q, v
  BN att_bn = rdBN(); Conv att_c = rdCv();
  const float* att_gamma = nxt();
  Conv att_k = rdCv(); Conv att_q = rdCv(); Conv att_v = rdCv();
  // params['conv1']
  Conv conv1 = rdCv();
  // params['down']
  BN down_bn = rdBN(); Conv down_c = rdCv();
  // params['l1_1'][6] : bn1, bn2, c1, c2, (sc)
  CNp l11[6];
  for (int i = 0; i < 6; ++i) {
    l11[i].bn1 = rdBN(); l11[i].bn2 = rdBN();
    l11[i].c1 = rdCv(); l11[i].c2 = rdCv();
    l11[i].has_sc = (i == 0);
    if (l11[i].has_sc) l11[i].sc = rdCv();
  }
  // params['l1_2'][6]
  CNp l12[6];
  for (int i = 0; i < 6; ++i) {
    l12[i].bn1 = rdBN(); l12[i].bn2 = rdBN();
    l12[i].c1 = rdCv(); l12[i].c2 = rdCv();
    l12[i].has_sc = (i == 0);
    if (l12[i].has_sc) l12[i].sc = rdCv();
  }
  // params['l2'][6] : bn1, bn2, bn3, c1, c2, c3
  OAp l2[6];
  for (int i = 0; i < 6; ++i) {
    l2[i].bn1 = rdBN(); l2[i].bn2 = rdBN(); l2[i].bn3 = rdBN();
    l2[i].c1 = rdCv(); l2[i].c2 = rdCv(); l2[i].c3 = rdCv();
  }
  // params['out'], params['s1'], params['up']
  Conv out_c = rdCv();
  BN s1_bn = rdBN(); Conv s1_c = rdCv();
  BN up_bn = rdBN(); Conv up_c = rdCv();

  // ---------------- workspace bump allocator --------------------------------
  char* wp = (char*)d_ws;
  auto allocF = [&](size_t nf) {
    float* p = (float*)wp;
    wp += ((nf * sizeof(float)) + 255) / 256 * 256;
    return p;
  };
  Stats st;
  st.imean = allocF(Bn * CL); st.irstd = allocF(Bn * CL);
  st.cmean = allocF(CL);      st.crstd = allocF(CL);

  float* x      = allocF((size_t)Bn * HALF * N);
  float* xsq    = allocF((size_t)Bn * N);
  float* G      = allocF((size_t)N * N);
  int*   idx    = (int*)allocF((size_t)Bn * N * Kk);
  float* feat   = allocF((size_t)Bn * C * NSP);        // graph features [B,128,40000]
  float* Rbig   = allocF((size_t)4 * Bn * HALF * NSP); // xl/q/k/v region; reused for x_l11
  float* xl = Rbig;
  float* qb = Rbig + (size_t)Bn * HALF * NSP;
  float* kb = Rbig + (size_t)2 * Bn * HALF * NSP;
  float* vb = Rbig + (size_t)3 * Bn * HALF * NSP;
  float* xsdg1 = allocF((size_t)Bn * HALF * N);
  float* xsdg2 = allocF((size_t)Bn * C * N);
  float* h192  = allocF((size_t)Bn * 192 * N);
  float* hA    = allocF((size_t)Bn * C * N);
  float* hB    = allocF((size_t)Bn * C * N);
  float* cn_t  = allocF((size_t)Bn * 256 * N);
  float* cn_h1 = allocF((size_t)Bn * C * N);
  float* cn_h2 = allocF((size_t)Bn * C * N);
  float* cn_sc = allocF((size_t)Bn * C * N);
  float* embed = allocF((size_t)Bn * CL * N);
  float* z     = allocF((size_t)Bn * C * CL);
  float* oa_t   = allocF((size_t)Bn * C * CL);
  float* oa_h1  = allocF((size_t)Bn * C * CL);
  float* oa_tr  = allocF((size_t)Bn * CL * C);
  float* oa_trn = allocF((size_t)Bn * CL * C);
  float* oa_h2  = allocF((size_t)Bn * CL * C);
  float* oa_h3  = allocF((size_t)Bn * C * CL);
  float* xup   = allocF((size_t)Bn * C * N);
  float* o256  = allocF((size_t)Bn * 256 * N);
  float* oA    = allocF((size_t)Bn * C * N);
  float* oB    = allocF((size_t)Bn * C * N);
  float* xwx   = allocF((size_t)Bn * 81);

  float* logits = (float*)d_out;                 // [B*N]
  float* ehat   = (float*)d_out + (size_t)Bn * N;          // [B*9]
  float* resid  = (float*)d_out + (size_t)Bn * N + Bn * 9; // [B*N]

  // ---------------- 1. x = conv1(data)  (M=64,K=4) --------------------------
  gemm(conv1.w, data, conv1.b, x, HALF, 4, N, 4, N, N,
       0, (long)4 * N, (long)HALF * N, Bn, 0, 0, s);

  // ---------------- 2. graph feature of x + attention -----------------------
  graph_feature(x, feat, xsq, G, idx, Bn, HALF, N, Kk, s);
  norm_relu(feat, feat, att_bn.g, att_bn.b, Bn, C, NSP, 1, 1, st, s);
  gemm(att_c.w, feat, att_c.b, xl, HALF, C, NSP, C, NSP, NSP,
       0, (long)C * NSP, (long)HALF * NSP, Bn, 0, 0, s);
  gemm(att_q.w, xl, att_q.b, qb, HALF, HALF, NSP, HALF, NSP, NSP,
       0, (long)HALF * NSP, (long)HALF * NSP, Bn, 0, 0, s);
  gemm(att_k.w, xl, att_k.b, kb, HALF, HALF, NSP, HALF, NSP, NSP,
       0, (long)HALF * NSP, (long)HALF * NSP, Bn, 0, 0, s);
  gemm(att_v.w, xl, att_v.b, vb, HALF, HALF, NSP, HALF, NSP, NSP,
       0, (long)HALF * NSP, (long)HALF * NSP, Bn, 0, 0, s);
  {
    long tot = (long)Bn * HALF * N;
    attn_kernel<<<grid1d(tot), TPB, 0, s>>>(qb, kb, vb, x, att_gamma, xsdg1, Kk, tot);
  }

  // ---------------- 3. graph feature of xsdg1 + s1 conv + maxpool ------------
  graph_feature(xsdg1, feat, xsq, G, idx, Bn, HALF, N, Kk, s);
  norm_relu(feat, feat, s1_bn.g, s1_bn.b, Bn, C, NSP, 1, 1, st, s);
  float* xl11 = Rbig;   // reuse big region (feat no longer needed after gemm)
  gemm(s1_c.w, feat, s1_c.b, xl11, C, C, NSP, C, NSP, NSP,
       0, (long)C * NSP, (long)C * NSP, Bn, 0, 0, s);
  {
    long tot = (long)Bn * C * N;
    maxk_kernel<<<grid1d(tot), TPB, 0, s>>>(xl11, xsdg2, Kk, tot);
  }

  // ---------------- 4. h = concat([xsdg1, xsdg2]) -> l1_1 stack --------------
  {
    long t1 = (long)Bn * HALF * N;
    copy_chan_kernel<<<grid1d(t1), TPB, 0, s>>>(h192, xsdg1, 192, HALF, 0, N, t1);
    long t2 = (long)Bn * C * N;
    copy_chan_kernel<<<grid1d(t2), TPB, 0, s>>>(h192, xsdg2, 192, C, HALF, N, t2);
  }
  const float* hin = h192;
  {
    float* outs[2] = { hA, hB };
    for (int l = 0; l < 6; ++l) {
      float* o = outs[l & 1];
      context_norm_layer(hin, o, (l == 0) ? 192 : C, l11[l],
                         cn_t, cn_h1, cn_h2, cn_sc, st, Bn, N, s);
      hin = o;
    }
  }
  const float* hfin = hin;   // == hB

  // ---------------- 5. diff_pool -> z ---------------------------------------
  norm_relu(hfin, cn_t, down_bn.g, down_bn.b, Bn, C, N, 1, 1, st, s);
  gemm(down_c.w, cn_t, down_c.b, embed, CL, C, N, C, N, N,
       0, (long)C * N, (long)CL * N, Bn, 0, 0, s);
  softmax_row_kernel<<<Bn * CL, TPB, 0, s>>>(embed, N);
  // z[c][k'] = sum_n h[c][n] * S[k'][n]  (transB)
  gemm(hfin, embed, nullptr, z, C, N, CL, N, N, CL,
       (long)C * N, (long)CL * N, (long)C * CL, Bn, 0, 1, s);

  // ---------------- 6. l2 stack of OAFilters on z ----------------------------
  for (int l = 0; l < 6; ++l) {
    const OAp& p = l2[l];
    norm_relu(z, oa_t, p.bn1.g, p.bn1.b, Bn, C, CL, 1, 1, st, s);
    gemm(p.c1.w, oa_t, p.c1.b, oa_h1, C, C, CL, C, CL, CL,
         0, (long)C * CL, (long)C * CL, Bn, 0, 0, s);
    long tcs = (long)Bn * C * CL;
    transpose_kernel<<<grid1d(tcs), TPB, 0, s>>>(oa_h1, oa_tr, C, CL, tcs);       // -> [B,500,128]
    norm_relu(oa_tr, oa_trn, p.bn2.g, p.bn2.b, Bn, CL, C, 0, 1, st, s);           // bnorm+relu only
    gemm(p.c2.w, oa_trn, p.c2.b, oa_h2, CL, CL, C, CL, C, C,
         0, (long)CL * C, (long)CL * C, Bn, 0, 0, s);
    add_kernel<<<grid1d(tcs), TPB, 0, s>>>(oa_tr, oa_tr, oa_h2, tcs);
    transpose_kernel<<<grid1d(tcs), TPB, 0, s>>>(oa_tr, oa_h1, CL, C, tcs);       // -> [B,128,500]
    norm_relu(oa_h1, oa_t, p.bn3.g, p.bn3.b, Bn, C, CL, 1, 1, st, s);
    gemm(p.c3.w, oa_t, p.c3.b, oa_h3, C, C, CL, C, CL, CL,
         0, (long)C * CL, (long)C * CL, Bn, 0, 0, s);
    add_kernel<<<grid1d(tcs), TPB, 0, s>>>(z, z, oa_h3, tcs);
  }

  // ---------------- 7. diff_unpool -> x_up -----------------------------------
  norm_relu(hfin, cn_t, up_bn.g, up_bn.b, Bn, C, N, 1, 1, st, s);
  gemm(up_c.w, cn_t, up_c.b, embed, CL, C, N, C, N, N,
       0, (long)C * N, (long)CL * N, Bn, 0, 0, s);
  {
    long tot = (long)Bn * N;
    softmax_chan_kernel<<<grid1d(tot), TPB, 0, s>>>(embed, CL, N, tot);
  }
  gemm(z, embed, nullptr, xup, C, CL, N, CL, N, N,
       (long)C * CL, (long)CL * N, (long)C * N, Bn, 0, 0, s);

  // ---------------- 8. o = concat([h, x_up]) -> l1_2 stack -------------------
  {
    long t1 = (long)Bn * C * N;
    copy_chan_kernel<<<grid1d(t1), TPB, 0, s>>>(o256, hfin, 256, C, 0, N, t1);
    copy_chan_kernel<<<grid1d(t1), TPB, 0, s>>>(o256, xup, 256, C, C, N, t1);
  }
  const float* oin = o256;
  {
    float* outs[2] = { oA, oB };
    for (int l = 0; l < 6; ++l) {
      float* o = outs[l & 1];
      context_norm_layer(oin, o, (l == 0) ? 256 : C, l12[l],
                         cn_t, cn_h1, cn_h2, cn_sc, st, Bn, N, s);
      oin = o;
    }
  }

  // ---------------- 9. logits, weighted 8-point, residual --------------------
  {
    long tot = (long)Bn * N;
    logits_kernel<<<grid1d(tot), TPB, 0, s>>>(oin, out_c.w, out_c.b, logits, C, N, tot);
  }
  xwx_kernel<<<Bn, TPB, 0, s>>>(xs, logits, xwx, N);
  eigh9_kernel<<<1, 32, 0, s>>>(xwx, ehat, Bn);
  {
    long tot = (long)Bn * N;
    episym_kernel<<<grid1d(tot), TPB, 0, s>>>(xs, ehat, resid, N, tot);
  }
}
